// LGEncoder_31731218382890
// MI455X (gfx1250) — matrix operand order repaired and checked by reading, not compiled, not run on hardware
//
#include <hip/hip_runtime.h>
#include <hip/hip_bf16.h>

typedef __attribute__((ext_vector_type(16))) _Float16 v16h;
typedef __attribute__((ext_vector_type(8)))  _Float16 v8h;
typedef __attribute__((ext_vector_type(8)))  float    v8f;

#define D_   256
#define FF_  512
#define L_   5
#define G_   4096
#define OUT_ 256
#define PH_  128

// ---------------------------------------------------------------------------
// WMMA fragment loaders (ISA 7.12.2 16-bit layouts, wave32)
// ---------------------------------------------------------------------------
__device__ __forceinline__ v16h cat16(v8h a, v8h b) {
  v16h r;
#pragma unroll
  for (int i = 0; i < 8; i++) { r[i] = a[i]; r[i + 8] = b[i]; }
  return r;
}

// A fragment: 16x32 (MxK) f16 from row-major A (leading dim lda).
// lane 0-15: M=lane, holds K {0..7,16..23}; lane 16-31: M=lane-16, K {8..15,24..31}
__device__ __forceinline__ v16h load_a(const _Float16* __restrict__ A, int lda, int lane) {
  int m  = lane & 15;
  int kh = (lane >> 4) * 8;
  v8h lo = *(const v8h*)(A + (size_t)m * lda + kh);
  v8h hi = *(const v8h*)(A + (size_t)m * lda + kh + 16);
  return cat16(lo, hi);
}

// B fragment: 32x16 (KxN) f16 from BT stored [N][K] row-major (ldb = K).
// lane 0-15: N=lane, K {0..7,16..23}; lane 16-31: N=lane-16, K {8..15,24..31}
__device__ __forceinline__ v16h load_b(const _Float16* __restrict__ BT, int ldb, int lane) {
  int n  = lane & 15;
  int kb = (lane >> 4) * 8;
  v8h lo = *(const v8h*)(BT + (size_t)n * ldb + kb);
  v8h hi = *(const v8h*)(BT + (size_t)n * ldb + kb + 16);
  return cat16(lo, hi);
}

// ---------------------------------------------------------------------------
// Tiled WMMA GEMM: C[M,Ncl] = act(A[M,K] @ BT[Ncl,K]^T + bias)
// block = 256 threads = 8 waves; wave tile 32x32 (4 wmma accs); block tile 128x64
// ---------------------------------------------------------------------------
template <int RELU, int OUT16>
__global__ __launch_bounds__(256) void wmma_gemm(
    const _Float16* __restrict__ A, const _Float16* __restrict__ BT,
    const float* __restrict__ bias, float* __restrict__ Cf,
    _Float16* __restrict__ Ch, int M, int K, int Ncl) {
  int lane  = threadIdx.x & 31;
  int wave  = threadIdx.x >> 5;
  int mbase = blockIdx.x * 128 + (wave >> 1) * 32;
  int nbase = blockIdx.y * 64  + (wave &  1) * 32;

  v8f acc00 = {}, acc01 = {}, acc10 = {}, acc11 = {};
  for (int k = 0; k < K; k += 32) {
    v16h a0 = load_a(A + (size_t)mbase        * K + k, K, lane);
    v16h a1 = load_a(A + (size_t)(mbase + 16) * K + k, K, lane);
    v16h b0 = load_b(BT + (size_t)nbase        * K + k, K, lane);
    v16h b1 = load_b(BT + (size_t)(nbase + 16) * K + k, K, lane);
    acc00 = __builtin_amdgcn_wmma_f32_16x16x32_f16(false, a0, false, b0, (short)0, acc00, false, false);
    acc01 = __builtin_amdgcn_wmma_f32_16x16x32_f16(false, a0, false, b1, (short)0, acc01, false, false);
    acc10 = __builtin_amdgcn_wmma_f32_16x16x32_f16(false, a1, false, b0, (short)0, acc10, false, false);
    acc11 = __builtin_amdgcn_wmma_f32_16x16x32_f16(false, a1, false, b1, (short)0, acc11, false, false);
  }

  // D layout: lane 0-15 -> N=lane, rows v; lane 16-31 -> rows v+8
  int n0   = nbase + (lane & 15);
  int n1   = n0 + 16;
  int mrow = (lane >> 4) * 8;
  float bi0 = bias[n0];
  float bi1 = bias[n1];
#pragma unroll
  for (int v = 0; v < 8; v++) {
    int m0 = mbase + v + mrow;
    int m1 = m0 + 16;
    float vals[4] = {acc00[v] + bi0, acc01[v] + bi1, acc10[v] + bi0, acc11[v] + bi1};
    int   ms[4]   = {m0, m0, m1, m1};
    int   ns[4]   = {n0, n1, n0, n1};
#pragma unroll
    for (int t = 0; t < 4; t++) {
      float xv = vals[t];
      if (RELU) xv = xv > 0.f ? xv : 0.f;
      if (ms[t] < M) {
        size_t off = (size_t)ms[t] * Ncl + ns[t];
        if (OUT16) Ch[off] = (_Float16)xv;
        else       Cf[off] = xv;
      }
    }
  }
}

// ---------------------------------------------------------------------------
// Graph / elementwise kernels
// ---------------------------------------------------------------------------
__global__ void k_init_h(const int* __restrict__ x, const float* __restrict__ ae1,
                         const float* __restrict__ ae2, float* __restrict__ h, int N) {
  int n = blockIdx.x, d = threadIdx.x;
  if (n < N) h[(size_t)n * D_ + d] = ae1[(size_t)x[n * 2] * D_ + d] + ae2[(size_t)x[n * 2 + 1] * D_ + d];
}

__global__ void k_scatter(const float* __restrict__ h, const int* __restrict__ ei,
                          const int* __restrict__ ea, const float* __restrict__ ee1,
                          const float* __restrict__ ee2, float* __restrict__ agg, int E) {
  int e = blockIdx.x, d = threadIdx.x;
  if (e >= E) return;
  int src = ei[e], dst = ei[E + e];
  int a0 = ea[e * 2], a1 = ea[e * 2 + 1];
  float m = h[(size_t)src * D_ + d] + ee1[(size_t)a0 * D_ + d] + ee2[(size_t)a1 * D_ + d];
  atomicAdd(&agg[(size_t)dst * D_ + d], m);
}

__global__ void k_transpose(const float* __restrict__ W, _Float16* __restrict__ WT, int K, int Ncl) {
  size_t slice = (size_t)K * Ncl;
  size_t idx = (size_t)blockIdx.x * 256 + threadIdx.x;
  if (idx >= slice) return;
  const float* Ws = W + (size_t)blockIdx.y * slice;
  _Float16*    Ts = WT + (size_t)blockIdx.y * slice;
  int k = (int)(idx / Ncl), n = (int)(idx % Ncl);
  Ts[(size_t)n * K + k] = (_Float16)Ws[idx];
}

__global__ void k_f32_to_f16(const float* __restrict__ s, _Float16* __restrict__ d, size_t cnt) {
  size_t i = (size_t)blockIdx.x * 256 + threadIdx.x;
  if (i < cnt) d[i] = (_Float16)s[i];
}

__global__ void k_zero(float* __restrict__ p, size_t cnt) {
  size_t i = (size_t)blockIdx.x * 256 + threadIdx.x;
  if (i < cnt) p[i] = 0.f;
}

__global__ void k_bn_reduce(const float* __restrict__ hp, float* __restrict__ sums, int N) {
  int d = threadIdx.x;
  int r0 = blockIdx.x * 256;
  float s = 0.f, s2 = 0.f;
  for (int i = 0; i < 256; i++) {
    int r = r0 + i;
    if (r < N) {
      float v = hp[(size_t)r * D_ + d];
      s += v; s2 += v * v;
    }
  }
  atomicAdd(&sums[d], s);
  atomicAdd(&sums[D_ + d], s2);
}

__global__ void k_bn_apply(const float* __restrict__ hp, const float* __restrict__ sums,
                           const float* __restrict__ gam, const float* __restrict__ bet,
                           float* __restrict__ h, int N, int relu) {
  int n = blockIdx.x, d = threadIdx.x;
  if (n >= N) return;
  float inv  = 1.f / (float)N;
  float mean = sums[d] * inv;
  float var  = sums[D_ + d] * inv - mean * mean;
  float xv = (hp[(size_t)n * D_ + d] - mean) * rsqrtf(var + 1e-5f) * gam[d] + bet[d];
  if (relu) xv = fmaxf(xv, 0.f);
  h[(size_t)n * D_ + d] = xv;
}

__global__ void k_pool(const float* __restrict__ h, const int* __restrict__ batch,
                       float* __restrict__ pooled, float* __restrict__ counts, int N) {
  int n = blockIdx.x, d = threadIdx.x;
  if (n >= N) return;
  int g = batch[n];
  atomicAdd(&pooled[(size_t)g * D_ + d], h[(size_t)n * D_ + d]);
  if (d == 0) atomicAdd(&counts[g], 1.f);
}

__global__ void k_pool_div(const float* __restrict__ pooled, const float* __restrict__ counts,
                           _Float16* __restrict__ ph) {
  int g = blockIdx.x, d = threadIdx.x;
  float c = fmaxf(counts[g], 1.f);
  ph[(size_t)g * D_ + d] = (_Float16)(pooled[(size_t)g * D_ + d] / c);
}

template <int SOFTPLUS>
__global__ void k_dense(const float* __restrict__ X, const float* __restrict__ W,
                        const float* __restrict__ b, float* __restrict__ Y,
                        int M, int K, int Ncl) {
  int idx = blockIdx.x * blockDim.x + threadIdx.x;
  if (idx >= M * Ncl) return;
  int m = idx / Ncl, n = idx % Ncl;
  float s = b[n];
  for (int k = 0; k < K; k++) s += X[(size_t)m * K + k] * W[(size_t)k * Ncl + n];
  if (SOFTPLUS) s = (s > 20.f) ? s : log1pf(expf(s));
  Y[idx] = s;
}

// ---------------------------------------------------------------------------
extern "C" void kernel_launch(void* const* d_in, const int* in_sizes, int n_in,
                              void* d_out, int out_size, void* d_ws, size_t ws_size,
                              hipStream_t stream) {
  const int*   x     = (const int*)d_in[0];
  const int*   ei    = (const int*)d_in[1];
  const int*   ea    = (const int*)d_in[2];
  const int*   batch = (const int*)d_in[3];
  const float* ae1   = (const float*)d_in[4];
  const float* ae2   = (const float*)d_in[5];
  const float* ee1   = (const float*)d_in[6];
  const float* ee2   = (const float*)d_in[7];
  const float* W1    = (const float*)d_in[8];
  const float* b1    = (const float*)d_in[9];
  const float* W2    = (const float*)d_in[10];
  const float* b2    = (const float*)d_in[11];
  const float* gam   = (const float*)d_in[12];
  const float* bet   = (const float*)d_in[13];
  const float* lin_w = (const float*)d_in[14];
  const float* lin_b = (const float*)d_in[15];
  const float* pw1   = (const float*)d_in[16];
  const float* pb1   = (const float*)d_in[17];
  const float* pw2   = (const float*)d_in[18];
  const float* pb2   = (const float*)d_in[19];
  const float* pw3   = (const float*)d_in[20];
  const float* pb3   = (const float*)d_in[21];

  const int N    = in_sizes[3];
  const int E    = in_sizes[1] / 2;
  const int Npad = ((N + 127) / 128) * 128;

  // workspace layout
  char* ws = (char*)d_ws;
  size_t off = 0;
  auto alloc = [&](size_t bytes) -> char* {
    char* p = ws + off;
    off += (bytes + 255) & ~(size_t)255;
    return p;
  };
  float*    h      = (float*)alloc((size_t)Npad * D_ * 4);
  float*    agg    = (float*)alloc((size_t)Npad * D_ * 4);   // also pre-BN h
  _Float16* aggh   = (_Float16*)alloc((size_t)Npad * D_ * 2);
  _Float16* hid    = (_Float16*)alloc((size_t)Npad * FF_ * 2);
  _Float16* w1t    = (_Float16*)alloc((size_t)L_ * FF_ * D_ * 2);
  _Float16* w2t    = (_Float16*)alloc((size_t)L_ * D_ * FF_ * 2);
  _Float16* lint   = (_Float16*)alloc((size_t)OUT_ * D_ * 2);
  float*    bns    = (float*)alloc(2 * D_ * 4);
  float*    pooled = (float*)alloc((size_t)G_ * D_ * 4);
  float*    counts = (float*)alloc((size_t)G_ * 4);
  _Float16* ph     = (_Float16*)alloc((size_t)G_ * D_ * 2);
  float*    p1     = (float*)alloc((size_t)G_ * PH_ * 4);
  float*    p2     = (float*)alloc((size_t)G_ * PH_ * 4);

  // one-time weight convert+transpose to f16 [n][k]
  {
    dim3 g1((D_ * FF_ + 255) / 256, L_);
    k_transpose<<<g1, 256, 0, stream>>>(W1, w1t, D_, FF_);
    dim3 g2((FF_ * D_ + 255) / 256, L_);
    k_transpose<<<g2, 256, 0, stream>>>(W2, w2t, FF_, D_);
    dim3 g3((D_ * OUT_ + 255) / 256, 1);
    k_transpose<<<g3, 256, 0, stream>>>(lin_w, lint, D_, OUT_);
  }

  k_init_h<<<N, D_, 0, stream>>>(x, ae1, ae2, h, N);

  for (int l = 0; l < L_; l++) {
    // agg = h + scatter_add(h[src] + e, dst)
    hipMemcpyAsync(agg, h, (size_t)N * D_ * 4, hipMemcpyDeviceToDevice, stream);
    k_scatter<<<E, D_, 0, stream>>>(h, ei, ea, ee1 + (size_t)l * 5 * D_,
                                    ee2 + (size_t)l * 3 * D_, agg, E);
    k_f32_to_f16<<<(unsigned)(((size_t)N * D_ + 255) / 256), 256, 0, stream>>>(agg, aggh, (size_t)N * D_);

    // hid = relu(agg @ W1 + b1)  [N,512] f16 out
    dim3 gg1(Npad / 128, FF_ / 64);
    wmma_gemm<1, 1><<<gg1, 256, 0, stream>>>(aggh, w1t + (size_t)l * FF_ * D_,
                                             b1 + (size_t)l * FF_, nullptr, hid, N, D_, FF_);
    // h_pre = hid @ W2 + b2     [N,256] f32 out (reuses agg)
    dim3 gg2(Npad / 128, D_ / 64);
    wmma_gemm<0, 0><<<gg2, 256, 0, stream>>>(hid, w2t + (size_t)l * D_ * FF_,
                                             b2 + (size_t)l * D_, agg, nullptr, N, FF_, D_);
    // batch norm over nodes
    k_zero<<<2, 256, 0, stream>>>(bns, (size_t)2 * D_);
    k_bn_reduce<<<(N + 255) / 256, D_, 0, stream>>>(agg, bns, N);
    k_bn_apply<<<N, D_, 0, stream>>>(agg, bns, gam + (size_t)l * D_, bet + (size_t)l * D_,
                                     h, N, l < L_ - 1);
  }

  // global mean pool
  k_zero<<<(G_ * D_ + 255) / 256, 256, 0, stream>>>(pooled, (size_t)G_ * D_);
  k_zero<<<(G_ + 255) / 256, 256, 0, stream>>>(counts, (size_t)G_);
  k_pool<<<N, D_, 0, stream>>>(h, batch, pooled, counts, N);
  k_pool_div<<<G_, D_, 0, stream>>>(pooled, counts, ph);

  // feat = pooled @ lin_w + lin_b  -> d_out[0 : G*OUT]
  float* feat = (float*)d_out;
  float* pred = (float*)d_out + (size_t)G_ * OUT_;
  dim3 gf(G_ / 128, OUT_ / 64);
  wmma_gemm<0, 0><<<gf, 256, 0, stream>>>(ph, lint, lin_b, feat, nullptr, G_, D_, OUT_);

  // prediction head (tiny; VALU dense)
  k_dense<1><<<(G_ * PH_ + 255) / 256, 256, 0, stream>>>(feat, pw1, pb1, p1, G_, OUT_, PH_);
  k_dense<1><<<(G_ * PH_ + 255) / 256, 256, 0, stream>>>(p1, pw2, pb2, p2, G_, PH_, PH_);
  k_dense<0><<<(G_ * 2 + 255) / 256, 256, 0, stream>>>(p2, pw3, pb3, pred, G_, PH_, 2);
}
